// ScagRcsmFusion_40329742909523
// MI455X (gfx1250) — matrix operand order repaired and checked by reading, not compiled, not run on hardware
//
#include <hip/hip_runtime.h>
#include <hip/hip_bf16.h>
#include <math.h>

typedef __attribute__((ext_vector_type(16))) _Float16 v16h;
typedef __attribute__((ext_vector_type(8)))  _Float16 v8h;
typedef __attribute__((ext_vector_type(8)))  float    v8f;

#define B_SZ     512
#define D_MODEL  768
#define OUT_DIM  256
#define D_INNER  1536
#define D_STATE  16
#define DT_RANK  48
#define DT_PAD   64
#define L_SEQ    4
#define M_ROWS   (B_SZ * L_SEQ)

// ---------------------------------------------------------------- helpers
__device__ __forceinline__ float actf(float x, int act) {
  if (act == 1) return 0.5f * x * (1.0f + erff(x * 0.7071067811865475f)); // exact gelu
  if (act == 2) return (x > 20.0f) ? x : log1pf(__expf(x));               // softplus
  return x;
}

__device__ __forceinline__ float block_reduce(float v, float* red) {
  const int tid = threadIdx.x;
  red[tid] = v;
  __syncthreads();
  for (int s = blockDim.x >> 1; s > 0; s >>= 1) {
    if (tid < s) red[tid] += red[tid + s];
    __syncthreads();
  }
  float r = red[0];
  __syncthreads();
  return r;
}

// ---------------------------------------------------------------- WMMA GEMM
// D[M x N] = act( A[M x Kp] * W[N x Kp]^T + bias )  ; f16 in, f32 accum.
// One wave -> 32x32 output (2 M-tiles x 2 N-tiles, 4 accumulators) so each
// A/B fragment is reused twice: ~16 flops per cache byte.
// A/B fragments both use contiguous k order (k = 16*half + e), not the ISA table order.
__global__ __launch_bounds__(128) void gemm_wmma(
    const _Float16* __restrict__ A, const _Float16* __restrict__ W,
    const float* __restrict__ bias, float* __restrict__ D32,
    _Float16* __restrict__ D16, int M, int N, int K, int act) {
  const int lane = threadIdx.x & 31;
  const int wave = threadIdx.x >> 5;
  const int nt0  = (blockIdx.x * 4 + wave) * 2;   // first 16-wide N tile
  if (nt0 * 16 >= N) return;                      // wave-uniform: EXEC stays all-1s
  const bool hasN1 = (nt0 + 1) * 16 < N;          // wave-uniform predicate
  const int rowM0 = blockIdx.y * 32;
  const int mloc  = lane & 15;                    // A row / B col / D col within tile
  const int half  = lane >> 4;

  const _Float16* __restrict__ Ar0 = A + (size_t)(rowM0 + mloc) * K;
  const _Float16* __restrict__ Ar1 = Ar0 + (size_t)16 * K;
  const _Float16* __restrict__ Wr0 = W + (size_t)(nt0 * 16 + mloc) * K;
  const _Float16* __restrict__ Wr1 = hasN1 ? (Wr0 + (size_t)16 * K) : Wr0;

  v8f c00 = {}, c10 = {}, c01 = {}, c11 = {};
  for (int k0 = 0; k0 < K; k0 += 32) {
    // B 32x16: lane(half,n) elems e -> K = k0 + 16*half + e (contiguous 32B)
    v16h b0 = *(const v16h*)(Wr0 + k0 + 16 * half);
    v16h b1 = *(const v16h*)(Wr1 + k0 + 16 * half);
    // A 16x32: elems e -> K = k0 + 16*half + e (contiguous, same as B)
    v8h lo0 = *(const v8h*)(Ar0 + k0 + 16 * half);
    v8h hi0 = *(const v8h*)(Ar0 + k0 + 16 * half + 8);
    v8h lo1 = *(const v8h*)(Ar1 + k0 + 16 * half);
    v8h hi1 = *(const v8h*)(Ar1 + k0 + 16 * half + 8);
    v16h a0, a1;
#pragma unroll
    for (int i = 0; i < 8; ++i) { a0[i] = lo0[i]; a0[8 + i] = hi0[i];
                                  a1[i] = lo1[i]; a1[8 + i] = hi1[i]; }
    c00 = __builtin_amdgcn_wmma_f32_16x16x32_f16(false, a0, false, b0, (short)0, c00, false, false);
    c10 = __builtin_amdgcn_wmma_f32_16x16x32_f16(false, a1, false, b0, (short)0, c10, false, false);
    c01 = __builtin_amdgcn_wmma_f32_16x16x32_f16(false, a0, false, b1, (short)0, c01, false, false);
    c11 = __builtin_amdgcn_wmma_f32_16x16x32_f16(false, a1, false, b1, (short)0, c11, false, false);
  }

  const int n0 = nt0 * 16 + mloc;
  const float bv0 = bias ? bias[n0] : 0.0f;
  const float bv1 = (bias && hasN1) ? bias[n0 + 16] : 0.0f;
#pragma unroll
  for (int r = 0; r < 8; ++r) {
    const int m0 = rowM0 + r + 8 * half;          // C/D layout: VGPR r -> row r + 8*half
    const int m1 = m0 + 16;
    float v00 = actf(c00[r] + bv0, act);
    float v10 = actf(c10[r] + bv0, act);
    if (D32) { D32[(size_t)m0 * N + n0] = v00; D32[(size_t)m1 * N + n0] = v10; }
    if (D16) { D16[(size_t)m0 * N + n0] = (_Float16)v00;
               D16[(size_t)m1 * N + n0] = (_Float16)v10; }
    if (hasN1) {
      const int n1 = n0 + 16;
      float v01 = actf(c01[r] + bv1, act);
      float v11 = actf(c11[r] + bv1, act);
      if (D32) { D32[(size_t)m0 * N + n1] = v01; D32[(size_t)m1 * N + n1] = v11; }
      if (D16) { D16[(size_t)m0 * N + n1] = (_Float16)v01;
                 D16[(size_t)m1 * N + n1] = (_Float16)v11; }
    }
  }
}

// ---------------------------------------------------------------- converts
__global__ void cvt_f16_pad(const float* __restrict__ src, int src_ld,
                            _Float16* __restrict__ dst, int dst_ld,
                            int rows, int cols) {
  const size_t total = (size_t)rows * dst_ld;
  for (size_t i = (size_t)blockIdx.x * blockDim.x + threadIdx.x; i < total;
       i += (size_t)gridDim.x * blockDim.x) {
    const int r = (int)(i / dst_ld), c = (int)(i % dst_ld);
    dst[i] = (c < cols) ? (_Float16)src[(size_t)r * src_ld + c] : (_Float16)0.0f;
  }
}

__global__ void make_joint16(const float* __restrict__ a, const float* __restrict__ b,
                             _Float16* __restrict__ dst) {
  const size_t total = (size_t)B_SZ * 2 * D_MODEL;
  for (size_t i = (size_t)blockIdx.x * blockDim.x + threadIdx.x; i < total;
       i += (size_t)gridDim.x * blockDim.x) {
    const int r = (int)(i / (2 * D_MODEL)), c = (int)(i % (2 * D_MODEL));
    const float v = (c < D_MODEL) ? a[(size_t)r * D_MODEL + c]
                                  : b[(size_t)r * D_MODEL + (c - D_MODEL)];
    dst[i] = (_Float16)v;
  }
}

// ---------------------------------------------------------------- conflict MLP
// one block (128 thr) per batch row: Y = relu(LN(X @ W^T + b; gamma,beta))
__global__ __launch_bounds__(128) void dense_ln_relu(
    const float* __restrict__ X, int ldx, const float* __restrict__ W,
    const float* __restrict__ b, const float* __restrict__ gamma,
    const float* __restrict__ beta, float* __restrict__ Y, int ldy, int K, int N) {
  __shared__ float yv[384];
  __shared__ float red[128];
  const int row = blockIdx.x;
  float psum = 0.f, psq = 0.f;
  for (int j = threadIdx.x; j < N; j += 128) {
    float acc = b[j];
    for (int k = 0; k < K; ++k) acc += X[(size_t)row * ldx + k] * W[(size_t)j * K + k];
    yv[j] = acc; psum += acc; psq += acc * acc;
  }
  const float inv = 1.0f / (float)N;
  const float mean = block_reduce(psum, red) * inv;
  const float var  = block_reduce(psq, red) * inv - mean * mean;
  const float rstd = rsqrtf(var + 1e-5f);
  for (int j = threadIdx.x; j < N; j += 128) {
    float v = (yv[j] - mean) * rstd * gamma[j] + beta[j];
    Y[(size_t)row * ldy + j] = fmaxf(v, 0.0f);
  }
}

__global__ __launch_bounds__(256) void conflict_cw(
    const float* __restrict__ h2, const float* __restrict__ W,
    const float* __restrict__ b, const float* __restrict__ cs,
    float* __restrict__ cw) {
  const int row = blockIdx.x;
  const float base = 1.0f - cs[row];
#pragma unroll
  for (int i = 0; i < 3; ++i) {
    const int j = threadIdx.x + i * 256;
    float acc = b[j];
    for (int k = 0; k < 384; ++k) acc += h2[(size_t)row * 384 + k] * W[(size_t)j * 384 + k];
    float lg = fminf(10.0f, fmaxf(-10.0f, base + acc));
    cw[(size_t)row * D_MODEL + j] = 1.0f / (1.0f + __expf(-lg));
  }
}

// ---------------------------------------------------------------- gating + stack + LN
// one block per (b,t): builds s, s16, xn16 (fwd) and flipped xn16 (bwd)
__global__ __launch_bounds__(256) void gate_stack_ln(
    const float* __restrict__ img_id, const float* __restrict__ img_attr,
    const float* __restrict__ txt_id, const float* __restrict__ txt_attr,
    const float* __restrict__ raw, const float* __restrict__ cw,
    const float* __restrict__ lnw, const float* __restrict__ lnb,
    float* __restrict__ s, _Float16* __restrict__ s16,
    _Float16* __restrict__ xnf, _Float16* __restrict__ xnb) {
  __shared__ float red[256];
  const int b = blockIdx.x >> 2, t = blockIdx.x & 3;
  const size_t r  = (size_t)blockIdx.x;
  const size_t rf = (size_t)(b * 4 + (3 - t));
  float vals[3], psum = 0.f, psq = 0.f;
#pragma unroll
  for (int i = 0; i < 3; ++i) {
    const int c = threadIdx.x + i * 256;
    const size_t bi = (size_t)b * D_MODEL + c;
    float v;
    if (t == 0) {
      float g = 1.0f / (1.0f + __expf(-(raw[(size_t)b * 1536 + c] - raw[(size_t)b * 1536 + 768 + c])));
      v = 0.7f * (img_id[bi] * g * cw[bi]) + 0.3f * img_id[bi];
    } else if (t == 1) {
      float g = 1.0f / (1.0f + __expf(-(raw[(size_t)b * 1536 + 768 + c] - raw[(size_t)b * 1536 + c])));
      v = 0.7f * (img_attr[bi] * g * cw[bi]) + 0.3f * img_attr[bi];
    } else if (t == 2) v = txt_id[bi];
    else               v = txt_attr[bi];
    vals[i] = v; psum += v; psq += v * v;
  }
  const float mean = block_reduce(psum, red) * (1.0f / D_MODEL);
  const float var  = block_reduce(psq, red) * (1.0f / D_MODEL) - mean * mean;
  const float rstd = rsqrtf(var + 1e-5f);
#pragma unroll
  for (int i = 0; i < 3; ++i) {
    const int c = threadIdx.x + i * 256;
    const float xn = (vals[i] - mean) * rstd * lnw[c] + lnb[c];
    s[r * D_MODEL + c]   = vals[i];
    s16[r * D_MODEL + c] = (_Float16)vals[i];
    xnf[r * D_MODEL + c]  = (_Float16)xn;
    xnb[rf * D_MODEL + c] = (_Float16)xn;   // flipped sequence for backward mamba
  }
}

// ---------------------------------------------------------------- mamba pieces
__global__ __launch_bounds__(256) void conv_silu(
    const float* __restrict__ xz, const float* __restrict__ cwgt,
    const float* __restrict__ cb, float* __restrict__ xh, _Float16* __restrict__ xh16) {
  const int b = blockIdx.x;
  for (int c = threadIdx.x; c < D_INNER; c += 256) {
    const float w0 = cwgt[c * 4 + 0], w1 = cwgt[c * 4 + 1],
                w2 = cwgt[c * 4 + 2], w3 = cwgt[c * 4 + 3];
    const float bb = cb[c];
    const size_t base = (size_t)b * 4;
    const float x0 = xz[(base + 0) * 3072 + c], x1 = xz[(base + 1) * 3072 + c],
                x2 = xz[(base + 2) * 3072 + c], x3 = xz[(base + 3) * 3072 + c];
    float y[4];
    y[0] = x0 * w3 + bb;
    y[1] = x0 * w2 + x1 * w3 + bb;
    y[2] = x0 * w1 + x1 * w2 + x2 * w3 + bb;
    y[3] = x0 * w0 + x1 * w1 + x2 * w2 + x3 * w3 + bb;
#pragma unroll
    for (int t = 0; t < 4; ++t) {
      const float v = y[t] / (1.0f + __expf(-y[t]));   // silu
      xh[(base + t) * D_INNER + c]   = v;
      xh16[(base + t) * D_INNER + c] = (_Float16)v;
    }
  }
}

__global__ __launch_bounds__(256) void ssm_scan(
    const float* __restrict__ dt, const float* __restrict__ xdb,
    const float* __restrict__ Alog, const float* __restrict__ Dp,
    const float* __restrict__ xh, const float* __restrict__ xz,
    float* __restrict__ y, _Float16* __restrict__ y16) {
  const int i = blockIdx.x * blockDim.x + threadIdx.x;
  if (i >= B_SZ * D_INNER) return;
  const int b = i / D_INNER, c = i % D_INNER;
  float Av[D_STATE], h[D_STATE];
#pragma unroll
  for (int n = 0; n < D_STATE; ++n) { Av[n] = -__expf(Alog[(size_t)c * D_STATE + n]); h[n] = 0.f; }
  const float Dc = Dp[c];
#pragma unroll
  for (int t = 0; t < L_SEQ; ++t) {
    const size_t r = (size_t)(b * 4 + t);
    const float dtv = dt[r * D_INNER + c];
    const float xv  = xh[r * D_INNER + c];
    const float* Bs = xdb + r * 80 + 48;
    const float* Cs = xdb + r * 80 + 64;
    float acc = 0.f;
#pragma unroll
    for (int n = 0; n < D_STATE; ++n) {
      h[n] = __expf(dtv * Av[n]) * h[n] + dtv * Bs[n] * xv;
      acc += h[n] * Cs[n];
    }
    const float zv = xz[r * 3072 + D_INNER + c];
    const float out = (acc + Dc * xv) * (zv / (1.0f + __expf(-zv)));
    y[r * D_INNER + c]   = out;
    y16[r * D_INNER + c] = (_Float16)out;
  }
}

// ---------------------------------------------------------------- fusion / head
__global__ __launch_bounds__(256) void fuse_ln(
    const float* __restrict__ s, const float* __restrict__ sf,
    const float* __restrict__ sb, const float* __restrict__ shrt,
    _Float16* __restrict__ out16) {
  __shared__ float red[256];
  const int b = blockIdx.x >> 2, t = blockIdx.x & 3;
  const size_t r  = (size_t)blockIdx.x;
  const size_t rb = (size_t)(b * 4 + (3 - t));      // un-flip backward mamba output
  float vals[3], psum = 0.f, psq = 0.f;
#pragma unroll
  for (int i = 0; i < 3; ++i) {
    const int c = threadIdx.x + i * 256;
    const float v = s[r * D_MODEL + c] + sf[r * D_MODEL + c] +
                    sb[rb * D_MODEL + c] + shrt[r * D_MODEL + c];
    vals[i] = v; psum += v; psq += v * v;
  }
  const float mean = block_reduce(psum, red) * (1.0f / D_MODEL);
  const float var  = block_reduce(psq, red) * (1.0f / D_MODEL) - mean * mean;
  const float rstd = rsqrtf(var + 1e-5f);
#pragma unroll
  for (int i = 0; i < 3; ++i) {
    const int c = threadIdx.x + i * 256;
    out16[r * D_MODEL + c] = (_Float16)((vals[i] - mean) * rstd);
  }
}

__global__ void mean_seq(const float* __restrict__ sout, float* __restrict__ f,
                         _Float16* __restrict__ f16) {
  const int i = blockIdx.x * blockDim.x + threadIdx.x;
  if (i >= B_SZ * D_MODEL) return;
  const int b = i / D_MODEL, c = i % D_MODEL;
  float a = 0.f;
#pragma unroll
  for (int t = 0; t < L_SEQ; ++t) a += sout[(size_t)(b * 4 + t) * D_MODEL + c];
  a *= 0.25f;
  f[i] = a; f16[i] = (_Float16)a;
}

__global__ __launch_bounds__(256) void head_final(
    const float* __restrict__ outlin, const float* __restrict__ rpout,
    const float* __restrict__ onw, const float* __restrict__ onb,
    float* __restrict__ dout) {
  __shared__ float red[256];
  const int b = blockIdx.x, j = threadIdx.x;
  const float v = outlin[(size_t)b * OUT_DIM + j];
  const float mean = block_reduce(v, red) * (1.0f / OUT_DIM);
  const float var  = block_reduce(v * v, red) * (1.0f / OUT_DIM) - mean * mean;
  const float ln = (v - mean) * rsqrtf(var + 1e-5f) * onw[j] + onb[j];
  dout[(size_t)b * OUT_DIM + j] = 0.7f * ln + 0.3f * rpout[(size_t)b * OUT_DIM + j];
}

// ---------------------------------------------------------------- launch
extern "C" void kernel_launch(void* const* d_in, const int* in_sizes, int n_in,
                              void* d_out, int out_size, void* d_ws, size_t ws_size,
                              hipStream_t stream) {
  (void)in_sizes; (void)n_in; (void)out_size; (void)ws_size;
  const float* img_id   = (const float*)d_in[0];
  const float* img_attr = (const float*)d_in[1];
  const float* txt_id   = (const float*)d_in[2];
  const float* txt_attr = (const float*)d_in[3];
  const float* cs       = (const float*)d_in[4];
  auto P = [&](int i) { return (const float*)d_in[i]; };
  // param indices (flattened dict order)
  // 5..14 conflict mlp, 15 mutual_w, 16 mutual_b, 17 ln_w, 18 ln_b
  // 19..27 mf, 28..36 mb, 37 inj_w, 38 inj_b, 39..42 bn, 43..48 head

  float* out_emb = (float*)d_out;
  float* cw      = out_emb + (size_t)B_SZ * OUT_DIM;

  char* base = (char*)d_ws; size_t off = 0;
  auto AL = [&](size_t bytes) -> void* {
    void* p = base + off; off += (bytes + 255) & ~(size_t)255; return p;
  };
  // f16 weights
  _Float16* Wmut = (_Float16*)AL((size_t)1536 * 1536 * 2);
  _Float16* WinF = (_Float16*)AL((size_t)3072 * 768 * 2);
  _Float16* WinB = (_Float16*)AL((size_t)3072 * 768 * 2);
  _Float16* WxF  = (_Float16*)AL((size_t)80 * 1536 * 2);
  _Float16* WxB  = (_Float16*)AL((size_t)80 * 1536 * 2);
  _Float16* WdtF = (_Float16*)AL((size_t)1536 * DT_PAD * 2);
  _Float16* WdtB = (_Float16*)AL((size_t)1536 * DT_PAD * 2);
  _Float16* WoF  = (_Float16*)AL((size_t)768 * 1536 * 2);
  _Float16* WoB  = (_Float16*)AL((size_t)768 * 1536 * 2);
  _Float16* Winj = (_Float16*)AL((size_t)768 * 768 * 2);
  _Float16* Wbn1 = (_Float16*)AL((size_t)3072 * 768 * 2);
  _Float16* Wbn2 = (_Float16*)AL((size_t)768 * 3072 * 2);
  _Float16* Wop  = (_Float16*)AL((size_t)256 * 768 * 2);
  _Float16* Wrp  = (_Float16*)AL((size_t)256 * 768 * 2);
  // activations
  float*    h1    = (float*)AL((size_t)B_SZ * 192 * 4);
  float*    h2    = (float*)AL((size_t)B_SZ * 384 * 4);
  _Float16* jnt16 = (_Float16*)AL((size_t)B_SZ * 1536 * 2);
  float*    raw   = (float*)AL((size_t)B_SZ * 1536 * 4);
  float*    sbufv = (float*)AL((size_t)M_ROWS * 768 * 4);
  _Float16* s16   = (_Float16*)AL((size_t)M_ROWS * 768 * 2);
  _Float16* xn16f = (_Float16*)AL((size_t)M_ROWS * 768 * 2);
  _Float16* xn16b = (_Float16*)AL((size_t)M_ROWS * 768 * 2);
  float*    xz    = (float*)AL((size_t)M_ROWS * 3072 * 4);
  float*    xh    = (float*)AL((size_t)M_ROWS * 1536 * 4);   // later aliased as hbn16
  _Float16* xh16  = (_Float16*)AL((size_t)M_ROWS * 1536 * 2);
  float*    xdb   = (float*)AL((size_t)M_ROWS * 80 * 4);
  _Float16* dti16 = (_Float16*)AL((size_t)M_ROWS * DT_PAD * 2);
  float*    dtb   = (float*)AL((size_t)M_ROWS * 1536 * 4);   // later aliased as s_out
  float*    ybuf  = (float*)AL((size_t)M_ROWS * 1536 * 4);
  _Float16* y16   = (_Float16*)AL((size_t)M_ROWS * 1536 * 2);
  float*    sfwd  = (float*)AL((size_t)M_ROWS * 768 * 4);
  float*    sbwd  = (float*)AL((size_t)M_ROWS * 768 * 4);
  float*    shrt  = (float*)AL((size_t)M_ROWS * 768 * 4);
  _Float16* sfu16 = (_Float16*)AL((size_t)M_ROWS * 768 * 2);
  float*    fusedv= (float*)AL((size_t)B_SZ * 768 * 4);
  _Float16* fus16 = (_Float16*)AL((size_t)B_SZ * 768 * 2);
  _Float16* iid16 = (_Float16*)AL((size_t)B_SZ * 768 * 2);
  float*    outln = (float*)AL((size_t)B_SZ * 256 * 4);
  float*    rpout = (float*)AL((size_t)B_SZ * 256 * 4);
  _Float16* hbn16 = (_Float16*)xh;   // alias: xh dead after scan (2048*3072*2 == 2048*1536*4)
  float*    s_out = dtb;             // alias: dt dead after scan

  auto CVT = [&](const float* src, int sld, _Float16* dst, int dld, int rows, int cols) {
    cvt_f16_pad<<<1024, 256, 0, stream>>>(src, sld, dst, dld, rows, cols);
  };
  auto GEMM = [&](const _Float16* Am, const _Float16* Wm, const float* bias,
                  float* D32, _Float16* D16, int M, int N, int K, int act) {
    dim3 g((N / 16 + 7) / 8, M / 32);              // wave = 32 rows x 32 cols
    gemm_wmma<<<g, 128, 0, stream>>>(Am, Wm, bias, D32, D16, M, N, K, act);
  };

  // --- weight conversion (f32 -> f16, dt_proj K padded 48->64)
  CVT(P(15), 1536, Wmut, 1536, 1536, 1536);
  CVT(P(19), 768,  WinF, 768,  3072, 768);
  CVT(P(28), 768,  WinB, 768,  3072, 768);
  CVT(P(22), 1536, WxF,  1536, 80,   1536);
  CVT(P(31), 1536, WxB,  1536, 80,   1536);
  CVT(P(23), 48,   WdtF, DT_PAD, 1536, 48);
  CVT(P(32), 48,   WdtB, DT_PAD, 1536, 48);
  CVT(P(27), 1536, WoF,  1536, 768,  1536);
  CVT(P(36), 1536, WoB,  1536, 768,  1536);
  CVT(P(37), 768,  Winj, 768,  768,  768);
  CVT(P(39), 768,  Wbn1, 768,  3072, 768);
  CVT(P(41), 3072, Wbn2, 3072, 768,  3072);
  CVT(P(43), 768,  Wop,  768,  256,  768);
  CVT(P(47), 768,  Wrp,  768,  256,  768);
  CVT(img_id, 768, iid16, 768, B_SZ, 768);
  make_joint16<<<1024, 256, 0, stream>>>(img_id, img_attr, jnt16);

  // --- conflict path -> cw (written directly into d_out tail)
  dense_ln_relu<<<B_SZ, 128, 0, stream>>>(cs, 1, P(5), P(6), P(7), P(8), h1, 192, 1, 192);
  dense_ln_relu<<<B_SZ, 128, 0, stream>>>(h1, 192, P(9), P(10), P(11), P(12), h2, 384, 192, 384);
  conflict_cw<<<B_SZ, 256, 0, stream>>>(h2, P(13), P(14), cs, cw);

  // --- mutual gating + stack + LN
  GEMM(jnt16, Wmut, P(16), raw, nullptr, B_SZ, 1536, 1536, 0);
  gate_stack_ln<<<M_ROWS, 256, 0, stream>>>(img_id, img_attr, txt_id, txt_attr,
                                            raw, cw, P(17), P(18),
                                            sbufv, s16, xn16f, xn16b);

  // --- two mamba passes (fwd, and bwd on flipped sequence)
  auto MAMBA = [&](const _Float16* xin, const _Float16* Win, const float* convw,
                   const float* convb, const _Float16* Wx, const _Float16* Wdt,
                   const float* dtbi, const float* Alog, const float* Dp,
                   const _Float16* Wo, float* sout) {
    GEMM(xin, Win, nullptr, xz, nullptr, M_ROWS, 3072, 768, 0);        // in_proj
    conv_silu<<<B_SZ, 256, 0, stream>>>(xz, convw, convb, xh, xh16);
    GEMM(xh16, Wx, nullptr, xdb, nullptr, M_ROWS, 80, 1536, 0);        // x_proj
    CVT(xdb, 80, dti16, DT_PAD, M_ROWS, 48);                            // dt slice, pad K
    GEMM(dti16, Wdt, dtbi, dtb, nullptr, M_ROWS, 1536, DT_PAD, 2);     // dt_proj + softplus
    ssm_scan<<<(B_SZ * D_INNER + 255) / 256, 256, 0, stream>>>(
        dtb, xdb, Alog, Dp, xh, xz, ybuf, y16);
    GEMM(y16, Wo, nullptr, sout, nullptr, M_ROWS, 768, 1536, 0);       // out_proj
  };
  MAMBA(xn16f, WinF, P(20), P(21), WxF, WdtF, P(24), P(25), P(26), WoF, sfwd);
  MAMBA(xn16b, WinB, P(29), P(30), WxB, WdtB, P(33), P(34), P(35), WoB, sbwd);

  // --- residual fuse + bottleneck + head
  GEMM(s16, Winj, P(38), shrt, nullptr, M_ROWS, 768, 768, 0);          // inj
  fuse_ln<<<M_ROWS, 256, 0, stream>>>(sbufv, sfwd, sbwd, shrt, sfu16);
  GEMM(sfu16, Wbn1, P(40), nullptr, hbn16, M_ROWS, 3072, 768, 1);      // bn1 + gelu -> f16
  GEMM(hbn16, Wbn2, P(42), s_out, nullptr, M_ROWS, 768, 3072, 0);      // bn2
  mean_seq<<<(B_SZ * 768 + 255) / 256, 256, 0, stream>>>(s_out, fusedv, fus16);
  GEMM(fus16, Wop, P(44), outln, nullptr, B_SZ, 256, 768, 0);          // op
  GEMM(iid16, Wrp, P(48), rpout, nullptr, B_SZ, 256, 768, 0);          // rp
  head_final<<<B_SZ, 256, 0, stream>>>(outln, rpout, P(45), P(46), out_emb);
}
